// MarginLinear_8349416424231
// MI455X (gfx1250) — matrix operand order repaired and checked by reading, not compiled, not run on hardware
//
#include <hip/hip_runtime.h>
#include <hip/hip_bf16.h>
#include <math.h>

// MI455X / gfx1250, wave32. Split-bf16 (hi+lo) WMMA GEMM: f32-grade accuracy at
// bf16 matrix-pipe rates (v_wmma_f32_16x16x32_bf16). Software-pipelined global
// loads; grid ordered so L2 (192MB) catches weight-tile reuse across M-blocks.

typedef __attribute__((ext_vector_type(16))) __bf16 v16bf;
typedef __attribute__((ext_vector_type(8)))  float  v8f;

#define BATCH   1024
#define FEAT    512
#define NUM_OUT 100000
#define LAMBDA_F (1000.0f / 1.12f)   // max(1000*(1+0.12*1)^-1, 5)

#define TM  256      // M rows per workgroup (1024 = 4*256, exact)
#define TN  80       // N cols per workgroup (100000 = 1250*80, exact)
#define KC  32       // K chunk = bf16 WMMA K
#define LDR 80       // LDS row stride bytes (32*2B = 64, padded to 5*16B -> conflict-free b128 reads)

// ---------------- Kernel A: row L2 norms (wave per row) ----------------
__global__ __launch_bounds__(256) void norms_kernel(const float* __restrict__ w,
                                                    const float* __restrict__ x,
                                                    float* __restrict__ invw,
                                                    float* __restrict__ xnorm) {
  const int wave = threadIdx.x >> 5;
  const int lane = threadIdx.x & 31;
  const int b = blockIdx.x;
  const float* src;
  int row;
  bool isW;
  if (b < 12500) { row = b * 8 + wave;           src = w + (size_t)row * FEAT; isW = true;  }
  else           { row = (b - 12500) * 8 + wave; src = x + (size_t)row * FEAT; isW = false; }

  const float4* v = (const float4*)src;   // 128 float4 per row
  float s = 0.0f;
#pragma unroll
  for (int j = 0; j < 4; ++j) {
    float4 t = v[lane + 32 * j];
    s += t.x * t.x + t.y * t.y + t.z * t.z + t.w * t.w;
  }
#pragma unroll
  for (int off = 16; off > 0; off >>= 1) s += __shfl_xor(s, off, 32);
  if (lane == 0) {
    if (isW) invw[row]  = 1.0f / sqrtf(s);
    else     xnorm[row] = sqrtf(s);
  }
}

// split f32 -> (hi, lo) bf16 pair and store 4 of each as 8B LDS writes
__device__ __forceinline__ void cvt_store4(unsigned char* dhi, unsigned char* dlo, float4 f) {
  union { __bf16 h[4]; uint2 u; } hi, lo;
  float fv[4] = { f.x, f.y, f.z, f.w };
#pragma unroll
  for (int j = 0; j < 4; ++j) {
    __bf16 h = (__bf16)fv[j];
    hi.h[j] = h;
    lo.h[j] = (__bf16)(fv[j] - (float)h);
  }
  *(uint2*)dhi = hi.u;
  *(uint2*)dlo = lo.u;
}

// ---------------- Kernel B: WMMA GEMM with split-bf16 ----------------
__global__ __launch_bounds__(256) void gemm_kernel(const float* __restrict__ x,
                                                   const float* __restrict__ w,
                                                   const float* __restrict__ invw,
                                                   float* __restrict__ out) {
  __shared__ __align__(16) unsigned char sAhi[TM * LDR];  // 20480 B
  __shared__ __align__(16) unsigned char sAlo[TM * LDR];
  __shared__ __align__(16) unsigned char sBhi[TN * LDR];  // 6400 B
  __shared__ __align__(16) unsigned char sBlo[TN * LDR];

  const int tid  = threadIdx.x;
  const int wave = tid >> 5;
  const int lane = tid & 31;
  const int half = lane >> 4;
  const int l16  = lane & 15;
  const int mblk = blockIdx.x;   // 0..3    (fast-varying -> adjacent blocks share w tile in L2)
  const int nblk = blockIdx.y;   // 0..1249

  v8f acc[10] = {};   // 2 M-tiles x 5 N-tiles of 16x16 per wave (wave owns 32x80)

  const float* xbase = x + (size_t)mblk * TM * FEAT;
  const float* wbase = w + (size_t)nblk * TN * FEAT;

  // per-thread staging coordinates (A: 2048 float4 / 256 thr = 8; B: 640 / 256 -> 3 w/ guard)
  const int rA[8] = { (tid + 0*256) >> 3, (tid + 1*256) >> 3, (tid + 2*256) >> 3, (tid + 3*256) >> 3,
                      (tid + 4*256) >> 3, (tid + 5*256) >> 3, (tid + 6*256) >> 3, (tid + 7*256) >> 3 };
  const int kqA = tid & 7;                 // same for all p (idx = tid + 256p, 256 % 8 == 0)
  const int rB[3] = { (tid + 0*256) >> 3, (tid + 1*256) >> 3, (tid + 2*256) >> 3 };
  const bool bOK = (tid + 2*256) < 640;    // third B slot only for half the threads

  float4 ra[8], rb[3];
  // -------- prologue: fetch k-chunk 0 --------
#pragma unroll
  for (int p = 0; p < 8; ++p)
    ra[p] = *(const float4*)(xbase + (size_t)rA[p] * FEAT + kqA * 4);
#pragma unroll
  for (int p = 0; p < 2; ++p)
    rb[p] = *(const float4*)(wbase + (size_t)rB[p] * FEAT + kqA * 4);
  if (bOK) rb[2] = *(const float4*)(wbase + (size_t)rB[2] * FEAT + kqA * 4);

#pragma unroll 1
  for (int kc = 0; kc < FEAT; kc += KC) {
    // -------- convert current chunk into LDS --------
#pragma unroll
    for (int p = 0; p < 8; ++p)
      cvt_store4(sAhi + rA[p] * LDR + kqA * 8, sAlo + rA[p] * LDR + kqA * 8, ra[p]);
#pragma unroll
    for (int p = 0; p < 2; ++p)
      cvt_store4(sBhi + rB[p] * LDR + kqA * 8, sBlo + rB[p] * LDR + kqA * 8, rb[p]);
    if (bOK)
      cvt_store4(sBhi + rB[2] * LDR + kqA * 8, sBlo + rB[2] * LDR + kqA * 8, rb[2]);
    __syncthreads();

    // -------- prefetch next chunk (latency hides under WMMAs) --------
    if (kc + KC < FEAT) {
      const int kn = kc + KC;
#pragma unroll
      for (int p = 0; p < 8; ++p)
        ra[p] = *(const float4*)(xbase + (size_t)rA[p] * FEAT + kn + kqA * 4);
#pragma unroll
      for (int p = 0; p < 2; ++p)
        rb[p] = *(const float4*)(wbase + (size_t)rB[p] * FEAT + kn + kqA * 4);
      if (bOK) rb[2] = *(const float4*)(wbase + (size_t)rB[2] * FEAT + kn + kqA * 4);
    }

    // -------- A fragments: 2 M-tiles (ISA 16-bit A layout) --------
    union Frag { v16bf v; uint4 q[2]; } ahi[2], alo[2], bhi, blo;
#pragma unroll
    for (int mt = 0; mt < 2; ++mt) {
      const unsigned char* ph = sAhi + (wave * 32 + mt * 16 + l16) * LDR + half * 16;
      const unsigned char* pl = sAlo + (wave * 32 + mt * 16 + l16) * LDR + half * 16;
      ahi[mt].q[0] = *(const uint4*)(ph);        // K 0..7  / 8..15  by half
      ahi[mt].q[1] = *(const uint4*)(ph + 32);   // K 16..23 / 24..31
      alo[mt].q[0] = *(const uint4*)(pl);
      alo[mt].q[1] = *(const uint4*)(pl + 32);
    }

#pragma unroll
    for (int t = 0; t < 5; ++t) {
      const unsigned char* pbh = sBhi + (t * 16 + l16) * LDR + half * 16;
      const unsigned char* pbl = sBlo + (t * 16 + l16) * LDR + half * 16;
      bhi.q[0] = *(const uint4*)(pbh);
      bhi.q[1] = *(const uint4*)(pbh + 32);
      blo.q[0] = *(const uint4*)(pbl);
      blo.q[1] = *(const uint4*)(pbl + 32);
#pragma unroll
      for (int mt = 0; mt < 2; ++mt) {
        v8f c = acc[mt * 5 + t];
        c = __builtin_amdgcn_wmma_f32_16x16x32_bf16(false, ahi[mt].v, false, bhi.v,
                                                    (short)0, c, false, false);
        c = __builtin_amdgcn_wmma_f32_16x16x32_bf16(false, alo[mt].v, false, bhi.v,
                                                    (short)0, c, false, false);
        c = __builtin_amdgcn_wmma_f32_16x16x32_bf16(false, ahi[mt].v, false, blo.v,
                                                    (short)0, c, false, false);
        acc[mt * 5 + t] = c;
      }
    }
    __syncthreads();
  }

  // -------- epilogue: scale by 1/||w_c||, store (D: M = i + 8*half, N = l16) --------
#pragma unroll
  for (int t = 0; t < 5; ++t) {
    const int col = nblk * TN + t * 16 + l16;
    const float sc = invw[col];
#pragma unroll
    for (int mt = 0; mt < 2; ++mt) {
      const int rowBase = mblk * TM + wave * 32 + mt * 16 + half * 8;
#pragma unroll
      for (int i = 0; i < 8; ++i) {
        out[(size_t)(rowBase + i) * NUM_OUT + col] = acc[mt * 5 + t][i] * sc;
      }
    }
  }
}

// ---------------- Kernel C: quadruple-angle margin fix-up ----------------
__device__ __forceinline__ float sgnf(float z) {
  return (z > 0.0f) ? 1.0f : ((z < 0.0f) ? -1.0f : 0.0f);
}

__global__ void margin_kernel(const int* __restrict__ target,
                              const float* __restrict__ xnorm,
                              float* __restrict__ out) {
  int n = blockIdx.x * blockDim.x + threadIdx.x;
  if (n >= BATCH) return;
  int c = target[n];
  size_t idx = (size_t)n * NUM_OUT + c;
  float v  = out[idx];           // xw[n, target[n]]
  float xn = xnorm[n];
  float ct = v / xn;
  float c2 = ct * ct;
  float c4 = c2 * c2;
  float s0 = sgnf(ct);
  float s3 = s0 * sgnf(2.0f * c2 - 1.0f);
  float s4 = 2.0f * s0 + s3 - 3.0f;
  float margin = xn * (s3 * (8.0f * c4 - 8.0f * c2 + 1.0f) + s4);
  out[idx] = (margin + LAMBDA_F * v) / (1.0f + LAMBDA_F);
}

extern "C" void kernel_launch(void* const* d_in, const int* in_sizes, int n_in,
                              void* d_out, int out_size, void* d_ws, size_t ws_size,
                              hipStream_t stream) {
  (void)in_sizes; (void)n_in; (void)out_size; (void)ws_size;
  const float* x      = (const float*)d_in[0];
  const int*   target = (const int*)d_in[1];
  const float* w      = (const float*)d_in[2];
  float* out = (float*)d_out;

  // workspace: [0,100000) inv weight-row norms, [100000,101024) x row norms
  float* invw  = (float*)d_ws;
  float* xnorm = invw + NUM_OUT;

  norms_kernel<<<12628, 256, 0, stream>>>(w, x, invw, xnorm);
  dim3 grid(4, 1250);   // x = M-block (fast) so all M-blocks of one w-tile are adjacent -> L2 reuse
  gemm_kernel<<<grid, 256, 0, stream>>>(x, w, invw, out);
  margin_kernel<<<4, 256, 0, stream>>>(target, xnorm, out);
}
